// Block_84859963834910
// MI455X (gfx1250) — compile-verified, matrix operand order checked by reading, not run on hardware
//
#include <hip/hip_runtime.h>
#include <hip/hip_bf16.h>

#define SEQ   3072
#define EMBED 1024
#define NHEAD 16
#define HDIM  64

typedef __attribute__((ext_vector_type(16))) __bf16 bf16x16;
typedef __attribute__((ext_vector_type(8)))  float  f32x8;

__device__ __forceinline__ unsigned short f2bf(float f) {
  union { float f; unsigned int u; } v; v.f = f;
  unsigned int r = v.u + 0x7fffu + ((v.u >> 16) & 1u);   // round-to-nearest-even
  return (unsigned short)(r >> 16);
}

// Build a 16xbf16 WMMA fragment from two contiguous 16-byte chunks
// (A-layout: K = {kb..kb+7} then {kb+16..kb+23}).
__device__ __forceinline__ bf16x16 frag_from(const unsigned short* p0,
                                             const unsigned short* p1) {
  uint4 t[2];
  t[0] = *(const uint4*)p0;
  t[1] = *(const uint4*)p1;
  bf16x16 v;
  __builtin_memcpy(&v, t, 32);
  return v;
}

__device__ __forceinline__ f32x8 wmma_bf16(bf16x16 a, bf16x16 b, f32x8 c) {
  return __builtin_amdgcn_wmma_f32_16x16x32_bf16(false, a, false, b,
                                                 (short)0, c, false, false);
}

// ---------------------------------------------------------------------------
// CDNA5 async copy: global -> LDS, 16B per lane, tracked by ASYNCcnt.
// Builtin wants pointers to a 16B int vector in AS1 (global) / AS3 (LDS).
// ---------------------------------------------------------------------------
typedef int v4i_t __attribute__((vector_size(16)));

__device__ __forceinline__ void async_b128(const void* g, void* l) {
#if __has_builtin(__builtin_amdgcn_global_load_async_to_lds_b128)
  __builtin_amdgcn_global_load_async_to_lds_b128(
      (__attribute__((address_space(1))) v4i_t*)(v4i_t*)(void*)g,
      (__attribute__((address_space(3))) v4i_t*)(v4i_t*)(void*)l, 0, 0);
#else
  unsigned lo = (unsigned)(unsigned long long)l;  // LDS byte offset (addr[31:0])
  asm volatile("global_load_async_to_lds_b128 %0, %1, off"
               :: "v"(lo), "v"(g) : "memory");
#endif
}

template <int N>
__device__ __forceinline__ void wait_asynccnt() {
#if __has_builtin(__builtin_amdgcn_s_wait_asynccnt)
  __builtin_amdgcn_s_wait_asynccnt(N);
#else
  if (N == 0) asm volatile("s_wait_asynccnt 0x0" ::: "memory");
  else        asm volatile("s_wait_asynccnt 0x4" ::: "memory");
#endif
}

// ---------------------------------------------------------------------------
// LayerNorm: one block per row, fp32 in -> bf16 out
// ---------------------------------------------------------------------------
__global__ void ln_kernel(const float* __restrict__ x, const float* __restrict__ w,
                          const float* __restrict__ b, unsigned short* __restrict__ out) {
  int row = blockIdx.x;
  const float* xr = x + (size_t)row * EMBED;
  int t = threadIdx.x;
  float v[4], s = 0.f, s2 = 0.f;
#pragma unroll
  for (int i = 0; i < 4; ++i) {
    v[i] = xr[t + i * 256];
    s += v[i]; s2 += v[i] * v[i];
  }
#pragma unroll
  for (int off = 16; off > 0; off >>= 1) {
    s  += __shfl_xor(s,  off, 32);
    s2 += __shfl_xor(s2, off, 32);
  }
  __shared__ float ls[8], ls2[8];
  int wid = t >> 5, lane = t & 31;
  if (lane == 0) { ls[wid] = s; ls2[wid] = s2; }
  __syncthreads();
  s = 0.f; s2 = 0.f;
#pragma unroll
  for (int i = 0; i < 8; ++i) { s += ls[i]; s2 += ls2[i]; }
  float mu  = s / (float)EMBED;
  float var = s2 / (float)EMBED - mu * mu;
  float inv = rsqrtf(var + 1e-5f);
  unsigned short* orow = out + (size_t)row * EMBED;
#pragma unroll
  for (int i = 0; i < 4; ++i) {
    int c = t + i * 256;
    orow[c] = f2bf((v[i] - mu) * inv * w[c] + b[c]);
  }
}

// ---------------------------------------------------------------------------
// Weight prep: W(f32)[K,N] -> Wt(bf16)[N,K], tiled 64x64 through LDS so both
// the global read and the global write are coalesced.
// ---------------------------------------------------------------------------
__global__ void wtrans_kernel(const float* __restrict__ W,
                              unsigned short* __restrict__ Wt, int K, int N) {
  __shared__ unsigned short tile[64 * 65];
  int k0 = blockIdx.y * 64, n0 = blockIdx.x * 64;
  int t = threadIdx.x;
#pragma unroll
  for (int i = 0; i < 16; ++i) {
    int idx = t + i * 256;
    int kk = idx >> 6, nn = idx & 63;
    tile[nn * 65 + kk] = f2bf(W[(size_t)(k0 + kk) * N + n0 + nn]);
  }
  __syncthreads();
#pragma unroll
  for (int i = 0; i < 16; ++i) {
    int idx = t + i * 256;
    int nn = idx >> 6, kk = idx & 63;
    Wt[(size_t)(n0 + nn) * K + k0 + kk] = tile[nn * 65 + kk];
  }
}

// ---------------------------------------------------------------------------
// GEMM: C[M,N] = A(bf16)[M,K] @ Bt(bf16)[N,K]^T + bias, mode epilogues.
// 256 threads (8 waves), tile 128x128x32, wave tile 64x32 (4x2 WMMA).
// Double-buffered LDS fed by async global->LDS b128 copies (ASYNCcnt pipeline):
// each thread issues 4 async b128 per stage (2 A rows + 2 B rows chunks).
// ---------------------------------------------------------------------------
#define BM 128
#define BN 128
#define BK 32

#define MODE_QKV       0   // scatter to q[h][S][64], k[h][S][64], vT[h][64][S] (bf16)
#define MODE_F32_RES   1   // out fp32 = acc + bias + residual
#define MODE_BF16_RELU 2   // out bf16 = relu(acc + bias)

__device__ __forceinline__ void stage_issue(const unsigned short* __restrict__ A,
                                            const unsigned short* __restrict__ Bt,
                                            unsigned short* As, unsigned short* Bs,
                                            int t, int bm0, int bn0, int K, int k0) {
#pragma unroll
  for (int i = 0; i < 2; ++i) {
    int j = t + i * 256;          // 512 16B transfers per tile
    int row = j >> 2;             // 0..127
    int ch  = (j & 3) * 8;        // element offset of 16B chunk within 32-elem row
    async_b128(A  + (size_t)(bm0 + row) * K + k0 + ch, As + row * BK + ch);
    async_b128(Bt + (size_t)(bn0 + row) * K + k0 + ch, Bs + row * BK + ch);
  }
}

template <int MODE>
__global__ void gemm_bf16_kernel(const unsigned short* __restrict__ A,
                                 const unsigned short* __restrict__ Bt,
                                 const float* __restrict__ bias,
                                 const float* __restrict__ residual,
                                 float* __restrict__ outF,
                                 unsigned short* __restrict__ outB,
                                 unsigned short* __restrict__ q,
                                 unsigned short* __restrict__ k,
                                 unsigned short* __restrict__ vT,
                                 int M, int N, int K) {
  __shared__ __align__(16) unsigned short As[2][BM * BK];
  __shared__ __align__(16) unsigned short Bs[2][BN * BK];

  int t = threadIdx.x;
  int lane = t & 31;
  int wid = t >> 5;            // 0..7
  int wm = wid & 1;            // M half (64 rows)
  int wn = wid >> 1;           // N quarter (32 cols)
  int bm0 = blockIdx.y * BM;
  int bn0 = blockIdx.x * BN;

  f32x8 acc[4][2];
#pragma unroll
  for (int mi = 0; mi < 4; ++mi)
#pragma unroll
    for (int ni = 0; ni < 2; ++ni)
      acc[mi][ni] = f32x8{0.f, 0.f, 0.f, 0.f, 0.f, 0.f, 0.f, 0.f};

  int kb = (lane >> 4) * 8;
  int cl = lane & 15;
  int nsteps = K / BK;

  stage_issue(A, Bt, As[0], Bs[0], t, bm0, bn0, K, 0);

  for (int s = 0; s < nsteps; ++s) {
    int cur = s & 1;
    if (s + 1 < nsteps) {
      stage_issue(A, Bt, As[cur ^ 1], Bs[cur ^ 1], t, bm0, bn0, K, (s + 1) * BK);
      wait_asynccnt<4>();      // stage s done (in-order), stage s+1 in flight
    } else {
      wait_asynccnt<0>();
    }
    __syncthreads();           // all waves' copies for stage s visible

    bf16x16 bf[2];
#pragma unroll
    for (int ni = 0; ni < 2; ++ni) {
      int col = wn * 32 + ni * 16 + cl;
      bf[ni] = frag_from(&Bs[cur][col * BK + kb], &Bs[cur][col * BK + kb + 16]);
    }
#pragma unroll
    for (int mi = 0; mi < 4; ++mi) {
      int row = wm * 64 + mi * 16 + cl;
      bf16x16 af = frag_from(&As[cur][row * BK + kb], &As[cur][row * BK + kb + 16]);
#pragma unroll
      for (int ni = 0; ni < 2; ++ni)
        acc[mi][ni] = wmma_bf16(af, bf[ni], acc[mi][ni]);
    }
    __syncthreads();           // done reading buf[cur] before it is re-filled
  }

  int halfAdd = (lane >> 4) * 8;
#pragma unroll
  for (int mi = 0; mi < 4; ++mi)
#pragma unroll
    for (int ni = 0; ni < 2; ++ni)
#pragma unroll
      for (int r = 0; r < 8; ++r) {
        int row = bm0 + wm * 64 + mi * 16 + r + halfAdd;
        int col = bn0 + wn * 32 + ni * 16 + cl;
        float v = acc[mi][ni][r] + bias[col];
        if constexpr (MODE == MODE_F32_RES) {
          outF[(size_t)row * N + col] = v + residual[(size_t)row * N + col];
        } else if constexpr (MODE == MODE_BF16_RELU) {
          outB[(size_t)row * N + col] = f2bf(v > 0.f ? v : 0.f);
        } else {  // MODE_QKV scatter
          if (col < EMBED) {
            int h = col >> 6, e = col & 63;
            q[((size_t)h * SEQ + row) * HDIM + e] = f2bf(v);
          } else if (col < 2 * EMBED) {
            int c = col - EMBED; int h = c >> 6, e = c & 63;
            k[((size_t)h * SEQ + row) * HDIM + e] = f2bf(v);
          } else {
            int c = col - 2 * EMBED; int h = c >> 6, e = c & 63;
            vT[((size_t)h * HDIM + e) * SEQ + row] = f2bf(v);  // transposed for P@V
          }
        }
      }
}

// ---------------------------------------------------------------------------
// Flash attention (causal + external mask), bf16 WMMA, online softmax.
// Block = 128 threads (4 waves); wave handles 16 query rows; key chunks of 32.
// ---------------------------------------------------------------------------
__global__ void attn_kernel(const unsigned short* __restrict__ qm,
                            const unsigned short* __restrict__ km,
                            const unsigned short* __restrict__ vT,
                            const unsigned char* __restrict__ mask,
                            unsigned short* __restrict__ out) {
  __shared__ __align__(16) unsigned short pls[4][16 * 32];  // per-wave P relayout

  int t = threadIdx.x;
  int lane = t & 31;
  int wid = t >> 5;
  int head = blockIdx.y;
  int q0 = blockIdx.x * 64 + wid * 16;

  const unsigned short* qh = qm + (size_t)head * SEQ * HDIM;
  const unsigned short* kh = km + (size_t)head * SEQ * HDIM;
  const unsigned short* vh = vT + (size_t)head * HDIM * SEQ;

  int cl = lane & 15;
  int kb = (lane >> 4) * 8;
  int halfAdd = (lane >> 4) * 8;

  // Q fragments: 16x64 = two 16x32 A-frags
  bf16x16 qa[2];
#pragma unroll
  for (int kc = 0; kc < 2; ++kc) {
    const unsigned short* p = qh + (size_t)(q0 + cl) * HDIM + kc * 32 + kb;
    qa[kc] = frag_from(p, p + 16);
  }

  f32x8 o[4];
#pragma unroll
  for (int ni = 0; ni < 4; ++ni)
    o[ni] = f32x8{0.f, 0.f, 0.f, 0.f, 0.f, 0.f, 0.f, 0.f};
  float m[8], l[8];
#pragma unroll
  for (int r = 0; r < 8; ++r) { m[r] = -3.0e38f; l[r] = 0.f; }

  for (int t0 = 0; t0 <= q0 + 15; t0 += 32) {  // causal early exit
    f32x8 s0 = f32x8{0.f, 0.f, 0.f, 0.f, 0.f, 0.f, 0.f, 0.f};
    f32x8 s1 = s0;
    // scores: q(16x64) @ k^T -> two 16x16 tiles (keys t0..+15, t0+16..+31)
#pragma unroll
    for (int kc = 0; kc < 2; ++kc) {
      const unsigned short* pk0 = kh + (size_t)(t0 + cl) * HDIM + kc * 32 + kb;
      const unsigned short* pk1 = kh + (size_t)(t0 + 16 + cl) * HDIM + kc * 32 + kb;
      s0 = wmma_bf16(qa[kc], frag_from(pk0, pk0 + 16), s0);
      s1 = wmma_bf16(qa[kc], frag_from(pk1, pk1 + 16), s1);
    }
    // scale + causal set + external mask add
    float sc0[8], sc1[8];
#pragma unroll
    for (int r = 0; r < 8; ++r) {
      int srow = q0 + r + halfAdd;
      int tc0 = t0 + cl, tc1 = t0 + 16 + cl;
      float a0 = s0[r] * 0.125f;   // 1/sqrt(64)
      float a1 = s1[r] * 0.125f;
      if (tc0 > srow) a0 = -10000.0f;
      if (tc1 > srow) a1 = -10000.0f;
      a0 += mask[tc0] ? 0.f : -10000.0f;
      a1 += mask[tc1] ? 0.f : -10000.0f;
      sc0[r] = a0; sc1[r] = a1;
    }
    // online softmax: row reductions within 16-lane groups (C-layout rows)
#pragma unroll
    for (int r = 0; r < 8; ++r) {
      float mx = fmaxf(sc0[r], sc1[r]);
#pragma unroll
      for (int off = 8; off > 0; off >>= 1) mx = fmaxf(mx, __shfl_xor(mx, off, 16));
      float mnew = fmaxf(m[r], mx);
      float rescale = __expf(m[r] - mnew);
      m[r] = mnew;
      float p0 = __expf(sc0[r] - mnew);
      float p1 = __expf(sc1[r] - mnew);
      sc0[r] = p0; sc1[r] = p1;
      float rs = p0 + p1;
#pragma unroll
      for (int off = 8; off > 0; off >>= 1) rs += __shfl_xor(rs, off, 16);
      l[r] = l[r] * rescale + rs;
#pragma unroll
      for (int ni = 0; ni < 4; ++ni) o[ni][r] *= rescale;
    }
    // P (C-layout) -> bf16 A-frag via per-wave LDS scratch (DS in-order per wave)
    unsigned short* pw = pls[wid];
#pragma unroll
    for (int r = 0; r < 8; ++r) {
      int rr = r + halfAdd;
      pw[rr * 32 + cl]      = f2bf(sc0[r]);
      pw[rr * 32 + 16 + cl] = f2bf(sc1[r]);
    }
    bf16x16 pa = frag_from(&pw[cl * 32 + kb], &pw[cl * 32 + kb + 16]);
    // o += P(16x32) @ V(32x64): V B-frags contiguous from vT[h][e][S]
#pragma unroll
    for (int ni = 0; ni < 4; ++ni) {
      const unsigned short* pv = vh + (size_t)(ni * 16 + cl) * SEQ + t0 + kb;
      o[ni] = wmma_bf16(pa, frag_from(pv, pv + 16), o[ni]);
    }
  }
  // normalize and store bf16 [S][EMBED]
#pragma unroll
  for (int ni = 0; ni < 4; ++ni)
#pragma unroll
    for (int r = 0; r < 8; ++r) {
      int row = q0 + r + halfAdd;
      int col = head * 64 + ni * 16 + cl;
      out[(size_t)row * EMBED + col] = f2bf(o[ni][r] / l[r]);
    }
}

// ---------------------------------------------------------------------------
extern "C" void kernel_launch(void* const* d_in, const int* in_sizes, int n_in,
                              void* d_out, int out_size, void* d_ws, size_t ws_size,
                              hipStream_t stream) {
  const float* hidden = (const float*)d_in[0];
  const unsigned char* mask = (const unsigned char*)d_in[1];
  const float* ln1_w  = (const float*)d_in[2];
  const float* ln1_b  = (const float*)d_in[3];
  const float* attn_w = (const float*)d_in[4];
  const float* attn_b = (const float*)d_in[5];
  const float* proj_w = (const float*)d_in[6];
  const float* proj_b = (const float*)d_in[7];
  const float* ln2_w  = (const float*)d_in[8];
  const float* ln2_b  = (const float*)d_in[9];
  const float* mlp_w1 = (const float*)d_in[10];
  const float* mlp_b1 = (const float*)d_in[11];
  const float* mlp_w2 = (const float*)d_in[12];
  const float* mlp_b2 = (const float*)d_in[13];
  float* outp = (float*)d_out;

  char* ws = (char*)d_ws;
  size_t off = 0;
  auto alloc = [&](size_t bytes) -> char* {
    char* p = ws + off;
    off += (bytes + 255) & ~(size_t)255;
    return p;
  };
  unsigned short* x_ln  = (unsigned short*)alloc((size_t)SEQ * EMBED * 2);
  unsigned short* qb    = (unsigned short*)alloc((size_t)SEQ * EMBED * 2);
  unsigned short* kbuf  = (unsigned short*)alloc((size_t)SEQ * EMBED * 2);
  unsigned short* vTb   = (unsigned short*)alloc((size_t)SEQ * EMBED * 2);
  unsigned short* aout  = (unsigned short*)alloc((size_t)SEQ * EMBED * 2);
  float*          hid2  = (float*)        alloc((size_t)SEQ * EMBED * 4);
  unsigned short* x2_ln = (unsigned short*)alloc((size_t)SEQ * EMBED * 2);
  unsigned short* mlp_a = (unsigned short*)alloc((size_t)SEQ * 4 * EMBED * 2);
  unsigned short* awT   = (unsigned short*)alloc((size_t)3 * EMBED * EMBED * 2);
  unsigned short* pwT   = (unsigned short*)alloc((size_t)EMBED * EMBED * 2);
  unsigned short* w1T   = (unsigned short*)alloc((size_t)4 * EMBED * EMBED * 2);
  unsigned short* w2T   = (unsigned short*)alloc((size_t)4 * EMBED * EMBED * 2);
  (void)ws_size; (void)in_sizes; (void)n_in; (void)out_size;

  // 0) weight prep: fp32 [K,N] -> bf16 transposed [N,K]
  wtrans_kernel<<<dim3(3 * EMBED / 64, EMBED / 64), 256, 0, stream>>>(attn_w, awT, EMBED, 3 * EMBED);
  wtrans_kernel<<<dim3(EMBED / 64, EMBED / 64), 256, 0, stream>>>(proj_w, pwT, EMBED, EMBED);
  wtrans_kernel<<<dim3(4 * EMBED / 64, EMBED / 64), 256, 0, stream>>>(mlp_w1, w1T, EMBED, 4 * EMBED);
  wtrans_kernel<<<dim3(EMBED / 64, 4 * EMBED / 64), 256, 0, stream>>>(mlp_w2, w2T, 4 * EMBED, EMBED);
  // 1) LN1 -> bf16
  ln_kernel<<<SEQ, 256, 0, stream>>>(hidden, ln1_w, ln1_b, x_ln);
  // 2) QKV GEMM, scatter epilogue to q/k/vT
  gemm_bf16_kernel<MODE_QKV><<<dim3(3 * EMBED / BN, SEQ / BM), 256, 0, stream>>>(
      x_ln, awT, attn_b, nullptr, nullptr, nullptr, qb, kbuf, vTb,
      SEQ, 3 * EMBED, EMBED);
  // 3) flash attention
  attn_kernel<<<dim3(SEQ / 64, NHEAD), 128, 0, stream>>>(qb, kbuf, vTb, mask, aout);
  // 4) proj GEMM + bias + residual(hidden) -> hid2 (fp32)
  gemm_bf16_kernel<MODE_F32_RES><<<dim3(EMBED / BN, SEQ / BM), 256, 0, stream>>>(
      aout, pwT, proj_b, hidden, hid2, nullptr, nullptr, nullptr, nullptr,
      SEQ, EMBED, EMBED);
  // 5) LN2 -> bf16
  ln_kernel<<<SEQ, 256, 0, stream>>>(hid2, ln2_w, ln2_b, x2_ln);
  // 6) MLP1 GEMM + bias + relu -> bf16
  gemm_bf16_kernel<MODE_BF16_RELU><<<dim3(4 * EMBED / BN, SEQ / BM), 256, 0, stream>>>(
      x2_ln, w1T, mlp_b1, nullptr, nullptr, mlp_a, nullptr, nullptr, nullptr,
      SEQ, 4 * EMBED, EMBED);
  // 7) MLP2 GEMM + bias + residual(hid2) -> out (fp32)
  gemm_bf16_kernel<MODE_F32_RES><<<dim3(EMBED / BN, SEQ / BM), 256, 0, stream>>>(
      mlp_a, w2T, mlp_b2, hid2, outp, nullptr, nullptr, nullptr, nullptr,
      SEQ, EMBED, 4 * EMBED);
}
